// HeterogeneousGraphNetwork_37220186587466
// MI455X (gfx1250) — matrix operand order repaired and checked by reading, not compiled, not run on hardware
//
#include <hip/hip_runtime.h>
#include <hip/hip_bf16.h>

// ---------------- problem constants ----------------
constexpr int B_    = 4;
constexpr int N_    = 1024;
constexpr int FD_   = 512;   // feature dim (== HID*HEADS)
constexpr int HID_  = 64;
constexpr int HEADS_= 8;

// ---------------- WMMA types / helpers ----------------
typedef __attribute__((ext_vector_type(16))) __bf16          v16bf;
typedef __attribute__((ext_vector_type(8)))  __bf16          v8bf;
typedef __attribute__((ext_vector_type(8)))  float           v8f;
typedef __attribute__((ext_vector_type(4)))  float           v4f;
typedef __attribute__((ext_vector_type(8)))  unsigned short  v8us;

// native f32 -> bf16 (RTE) via hardware cvt
__device__ inline unsigned short f2bf(float f) {
  union { __bf16 b; unsigned short u; } c;
  c.b = (__bf16)f;
  return c.u;
}
__device__ inline v8f zero8() {
  v8f z = {0.f,0.f,0.f,0.f,0.f,0.f,0.f,0.f};
  return z;
}
__device__ inline v8f wmma_bf16(v16bf a, v16bf b, v8f c) {
  return __builtin_amdgcn_wmma_f32_16x16x32_bf16(
      false, a, false, b, (short)0, c, false, false);
}

// 8-element (16B) chunk load, converting to bf16 if needed
__device__ inline v8us ld8_bf(const float* p) {
  v4f a = *(const v4f*)p;
  v4f b = *(const v4f*)(p + 4);
  v8us r;
#pragma unroll
  for (int i = 0; i < 4; ++i) { r[i] = f2bf(a[i]); r[4 + i] = f2bf(b[i]); }
  return r;
}
__device__ inline v8us ld8_bf(const unsigned short* p) {
  return *(const v8us*)p;
}

// Gather a fragment from two 16-byte LDS segments (each -> ds_load_b128)
__device__ inline v16bf ld_frag2(const unsigned short* p0,
                                 const unsigned short* p1) {
  v8bf lo = *(const v8bf*)p0;
  v8bf hi = *(const v8bf*)p1;
  v16bf r;
#pragma unroll
  for (int i = 0; i < 8; ++i) { r[i] = lo[i]; r[8 + i] = hi[i]; }
  return r;
}

// A-fragment: 16x32 bf16 tile, row-major in LDS (lda = row pitch, mult of 8).
__device__ inline v16bf load_a_frag(const unsigned short* lds, int lda) {
  const int lane = threadIdx.x & 31;
  const unsigned short* base = lds + (lane & 15) * lda + (lane >> 4) * 8;
  return ld_frag2(base, base + 16);
}
// B-fragment: 32x16 (KxN) tile stored TRANSPOSED in LDS: Bt[n][k], pitch ldt.
__device__ inline v16bf load_bt_frag(const unsigned short* ldsT, int ldt) {
  const int lane = threadIdx.x & 31;
  const unsigned short* base = ldsT + (lane & 15) * ldt + (lane >> 4) * 8;
  return ld_frag2(base, base + 16);
}

// =====================================================================
// Kernel 1: generic per-(b,h) GEMM  out[bz] = A(MxF) * W(FxKout=64) + bias
// Compile-time strides; vectorized 16B staging.
// 4 waves/block, 32 rows x 64 cols per block (2 WMMAs share one B-frag).
// =====================================================================
template <typename TA, int SAN, int LDW1, int LDW2>
__global__ void __launch_bounds__(128, 1) gemm_bf16_kernel(
    const TA* __restrict__ A, long long sAb, long long sAh,
    const float* __restrict__ W, long long sWh,
    const float* __restrict__ bias,
    unsigned short* __restrict__ out,
    int M, int Fdim, int H)
{
  __shared__ alignas(16) unsigned short As[32][40];   // A rows (bf16)
  __shared__ alignas(16) unsigned short Wt[64][40];   // B transposed: Wt[k][f]
  const int t    = threadIdx.x;
  const int w    = t >> 5, lane = t & 31, grp = lane >> 4, cl = lane & 15;
  const int bz   = blockIdx.z, b = bz / H, h = bz % H;
  const int row0 = blockIdx.x * 32;

  const TA*    Ab = A + (long long)b * sAb + (long long)h * sAh
                      + (long long)row0 * SAN;
  const float* Wb = W + (long long)h * sWh;

  v8f acc0 = zero8(), acc1 = zero8();
  for (int f0 = 0; f0 < Fdim; f0 += 32) {
    if (f0 + 32 < Fdim)   // gfx1250 global_prefetch path
      __builtin_prefetch(Ab + (t & 31) * SAN + f0 + 32, 0, 1);
    {   // A tile: 32 rows x 32 f = 128 chunks of 8 -> one per thread
      int r = t >> 2, c8 = (t & 3) * 8;
      *(v8us*)&As[r][c8] = ld8_bf(Ab + r * SAN + f0 + c8);
    }
    if constexpr (LDW2 == 1) {          // W rows contiguous in k
#pragma unroll
      for (int e = t; e < 32*8; e += 128) {
        int ci = e >> 3, k8 = (e & 7) * 8;
        v8us v = ld8_bf(Wb + (f0 + ci) * LDW1 + k8);
#pragma unroll
        for (int j = 0; j < 8; ++j) Wt[k8 + j][ci] = v[j];
      }
    } else {                            // strided source (ew^T): elementwise
#pragma unroll
      for (int e = t; e < 32*64; e += 128) {
        int ci = e >> 6, k = e & 63;
        Wt[k][ci] = f2bf(Wb[(f0 + ci) * LDW1 + k * LDW2]);
      }
    }
    __syncthreads();
    v16bf bfr = load_bt_frag(&Wt[w*16][0], 40);
    v16bf af0 = load_a_frag(&As[0][0],  40);
    v16bf af1 = load_a_frag(&As[16][0], 40);
    acc0 = wmma_bf16(af0, bfr, acc0);
    acc1 = wmma_bf16(af1, bfr, acc1);
    __syncthreads();
  }

  const int col = w*16 + cl;
  const float bv = bias ? bias[h * HID_ + col] : 0.0f;
  unsigned short* ob = out + (long long)bz * M * HID_ + row0 * HID_;
#pragma unroll
  for (int i = 0; i < 8; ++i) {
    int r = grp*8 + i;
    ob[r * HID_ + col]        = f2bf(acc0[i] + bv);
    ob[(r + 16) * HID_ + col] = f2bf(acc1[i] + bv);
  }
}

// =====================================================================
// Kernel 2: fused attention (flash style, no NxN materialization)
//   S = F1(16x64) @ G^T, S *= adj, online softmax, O += P @ F3
// Compile-time adjacency strides (SAN,SAM).  4 waves = 64 rows / block.
// =====================================================================
template <int SAN, int SAM>
__global__ void __launch_bounds__(128, 1) attn_kernel(
    const unsigned short* __restrict__ F1,
    const unsigned short* __restrict__ G,
    const unsigned short* __restrict__ F3,
    const float* __restrict__ adj,
    unsigned short* __restrict__ msgs, int H)
{
  __shared__ alignas(16) unsigned short F1s[64][72];      // rows (A of S)
  __shared__ alignas(16) unsigned short Gs [64][72];      // Bt of S = G rows
  __shared__ alignas(16) unsigned short F3t[64][72];      // Bt of O: F3t[k][m]
  __shared__ alignas(16) unsigned short Ps [4][16][40];   // P rows (A of O)

  const int t    = threadIdx.x;
  const int w    = t >> 5, lane = t & 31, grp = lane >> 4, cl = lane & 15;
  const int bz   = blockIdx.z, b = bz / H;
  const int row0 = blockIdx.x * 64;
  const int wr0  = w * 16;

  const unsigned short* F1b = F1 + (long long)bz * N_ * HID_;
  const unsigned short* Gb  = G  + (long long)bz * N_ * HID_;
  const unsigned short* F3b = F3 + (long long)bz * N_ * HID_;
  unsigned short*       Mb  = msgs + (long long)bz * N_ * HID_;
  const float* adjp = adj + (long long)b * N_ * N_
                          + (long long)(row0 + wr0) * SAN;

#pragma unroll
  for (int e = t; e < 64*8; e += 128) {       // 512 x 16B chunks
    int r = e >> 3, c8 = (e & 7) * 8;
    *(v8us*)&F1s[r][c8] = *(const v8us*)(F1b + (row0 + r) * HID_ + c8);
  }

  v8f  Oacc[4];
  float rmax[8], rsum[8];
#pragma unroll
  for (int ct = 0; ct < 4; ++ct) Oacc[ct] = zero8();
#pragma unroll
  for (int i = 0; i < 8; ++i) { rmax[i] = -1e30f; rsum[i] = 0.f; }

  for (int mb = 0; mb < N_/64; ++mb) {
    const int col0 = mb * 64;
#pragma unroll
    for (int e = t; e < 64*8; e += 128) {     // 16B-chunk staging
      int m = e >> 3, c8 = (e & 7) * 8;
      int src = (col0 + m) * HID_ + c8;
      *(v8us*)&Gs[m][c8] = *(const v8us*)(Gb + src);   // contiguous copy
      v8us f3v = *(const v8us*)(F3b + src);            // transpose scatter
#pragma unroll
      for (int j = 0; j < 8; ++j) F3t[c8 + j][m] = f3v[j];
    }
    __syncthreads();

    // S = F1 @ G^T   (16 x 64 per wave)
    v8f S[4];
#pragma unroll
    for (int ct = 0; ct < 4; ++ct) {
      S[ct] = zero8();
#pragma unroll
      for (int kc = 0; kc < 2; ++kc) {
        v16bf af  = load_a_frag (&F1s[wr0][kc*32],  72);
        v16bf bfr = load_bt_frag(&Gs[ct*16][kc*32], 72);
        S[ct] = wmma_bf16(af, bfr, S[ct]);
      }
    }

    // multiplicative adjacency mask + online softmax
    float nm[8];
#pragma unroll
    for (int i = 0; i < 8; ++i) nm[i] = rmax[i];
    {
      const float* ap = adjp + grp * (8 * SAN) + (col0 + cl) * SAM;
#pragma unroll
      for (int ct = 0; ct < 4; ++ct)
#pragma unroll
        for (int i = 0; i < 8; ++i) {
          float av = __builtin_nontemporal_load(ap + i * SAN + ct * (16 * SAM));
          float x  = S[ct][i] * av;
          S[ct][i] = x;
          nm[i] = fmaxf(nm[i], x);
        }
    }
    for (int d = 1; d < 16; d <<= 1)
#pragma unroll
      for (int i = 0; i < 8; ++i)
        nm[i] = fmaxf(nm[i], __shfl_xor(nm[i], d, 32));

    float sc[8], rs[8];
#pragma unroll
    for (int i = 0; i < 8; ++i) {
      sc[i] = __expf(rmax[i] - nm[i]);
      rmax[i] = nm[i];
      rs[i] = 0.f;
    }
#pragma unroll
    for (int ct = 0; ct < 4; ++ct)
#pragma unroll
      for (int i = 0; i < 8; ++i) {
        float p = __expf(S[ct][i] - nm[i]);
        rs[i] += p;
        Ps[w][grp*8 + i][ct*16 + cl] = f2bf(p);   // relayout C->A via LDS
      }
    for (int d = 1; d < 16; d <<= 1)
#pragma unroll
      for (int i = 0; i < 8; ++i)
        rs[i] += __shfl_xor(rs[i], d, 32);
#pragma unroll
    for (int i = 0; i < 8; ++i) rsum[i] = rsum[i]*sc[i] + rs[i];
#pragma unroll
    for (int ct = 0; ct < 4; ++ct)
#pragma unroll
      for (int i = 0; i < 8; ++i) Oacc[ct][i] *= sc[i];

    // O += P @ F3
#pragma unroll
    for (int ct = 0; ct < 4; ++ct)
#pragma unroll
      for (int mc = 0; mc < 2; ++mc) {
        v16bf af  = load_a_frag (&Ps[w][0][mc*32],   40);
        v16bf bfr = load_bt_frag(&F3t[ct*16][mc*32], 72);
        Oacc[ct] = wmma_bf16(af, bfr, Oacc[ct]);
      }
    __syncthreads();
  }

#pragma unroll
  for (int ct = 0; ct < 4; ++ct)
#pragma unroll
    for (int i = 0; i < 8; ++i) {
      int row = row0 + wr0 + grp*8 + i;
      float v = Oacc[ct][i] / rsum[i];
      Mb[row * HID_ + ct*16 + cl] = f2bf(v);
    }
}

// =====================================================================
// Kernel 3: output projection
//  out[b,n,f'] (+)= relu(concat(msgs)[b,n,:] @ Wout^T + b_out) + b_output
// 32 rows x 64 cols per block; vectorized staging.
// =====================================================================
__global__ void __launch_bounds__(128, 1) outproj_kernel(
    const unsigned short* __restrict__ msgs,
    const float* __restrict__ Wout,
    const float* __restrict__ b_out,
    const float* __restrict__ b_output,
    float* __restrict__ out, int accumulate)
{
  __shared__ alignas(16) unsigned short As[32][40];
  __shared__ alignas(16) unsigned short Wt[64][40];   // Wt[f'][c] (Bt layout)
  const int t    = threadIdx.x;
  const int w    = t >> 5, lane = t & 31, grp = lane >> 4, cl = lane & 15;
  const int b    = blockIdx.z;
  const int row0 = blockIdx.x * 32;
  const int f0   = blockIdx.y * 64;

  const unsigned short* Mb = msgs + (long long)b * HEADS_ * N_ * HID_;
  float* Ob = out + (long long)b * N_ * FD_ + row0 * FD_;

  v8f acc0 = zero8(), acc1 = zero8();
  for (int c0 = 0; c0 < FD_; c0 += 32) {
    const int h  = c0 >> 6;            // uniform per chunk (32 | 64)
    const int k0 = c0 & 63;
    if (c0 + 32 < FD_)
      __builtin_prefetch(Wout + (f0 + (t & 31)) * FD_ + c0 + 32, 0, 1);
    {   // A tile: 128 x 16B chunks, one per thread
      int r = t >> 2, c8 = (t & 3) * 8;
      *(v8us*)&As[r][c8] =
          *(const v8us*)(Mb + (h * N_ + row0 + r) * HID_ + k0 + c8);
    }
#pragma unroll
    for (int e = t; e < 64*4; e += 128) {     // Wt rows: 16B chunks
      int fj = e >> 2, c8 = (e & 3) * 8;
      *(v8us*)&Wt[fj][c8] = ld8_bf(Wout + (f0 + fj) * FD_ + c0 + c8);
    }
    __syncthreads();
    v16bf bfr = load_bt_frag(&Wt[w*16][0], 40);
    v16bf af0 = load_a_frag(&As[0][0],  40);
    v16bf af1 = load_a_frag(&As[16][0], 40);
    acc0 = wmma_bf16(af0, bfr, acc0);
    acc1 = wmma_bf16(af1, bfr, acc1);
    __syncthreads();
  }

  const int col = f0 + w*16 + cl;
  const float bo  = b_out[col];
  const float bo2 = b_output[col];
#pragma unroll
  for (int i = 0; i < 8; ++i) {
    int r = grp*8 + i;
    float v0 = fmaxf(acc0[i] + bo, 0.0f) + bo2;
    float v1 = fmaxf(acc1[i] + bo, 0.0f) + bo2;
    int i0 = r * FD_ + col;
    int i1 = (r + 16) * FD_ + col;
    if (accumulate) { Ob[i0] += v0; Ob[i1] += v1; }
    else            { Ob[i0]  = v0; Ob[i1]  = v1; }
  }
}

// =====================================================================
// Host launch
// =====================================================================
extern "C" void kernel_launch(void* const* d_in, const int* in_sizes, int n_in,
                              void* d_out, int out_size, void* d_ws, size_t ws_size,
                              hipStream_t stream) {
  (void)in_sizes; (void)n_in; (void)out_size; (void)ws_size;

  const float* fa   = (const float*)d_in[0];
  const float* fb   = (const float*)d_in[1];
  const float* Aia  = (const float*)d_in[2];
  const float* Aib  = (const float*)d_in[3];
  const float* Ait  = (const float*)d_in[4];
  const float* W1   = (const float*)d_in[5];  const float* b1 = (const float*)d_in[6];
  const float* W2   = (const float*)d_in[7];  const float* b2 = (const float*)d_in[8];
  const float* W3   = (const float*)d_in[9];  const float* b3 = (const float*)d_in[10];
  const float* Wia  = (const float*)d_in[11];
  const float* Wib  = (const float*)d_in[12];
  const float* Wit  = (const float*)d_in[13];
  const float* Wout = (const float*)d_in[14];
  const float* bout = (const float*)d_in[15];
  const float* boutp= (const float*)d_in[16];

  // bf16 workspace: 6 projections + G + msgs, each B*H*N*HID ushorts (4 MB)
  unsigned short* ws = (unsigned short*)d_ws;
  const size_t SZ = (size_t)B_ * HEADS_ * N_ * HID_;
  unsigned short* F1a = ws + 0*SZ;
  unsigned short* F2a = ws + 1*SZ;
  unsigned short* F3a = ws + 2*SZ;
  unsigned short* F1b = ws + 3*SZ;
  unsigned short* F2b = ws + 4*SZ;
  unsigned short* F3b = ws + 5*SZ;
  unsigned short* Gw  = ws + 6*SZ;
  unsigned short* Mw  = ws + 7*SZ;

  const dim3 blk(128);
  const dim3 gProj(N_/32, 1, B_*HEADS_);
  const dim3 gAttn(N_/64, 1, B_*HEADS_);
  const dim3 gOut (N_/32, FD_/64, B_);

  auto proj = [&](const float* X, const float* W, const float* bias,
                  unsigned short* outp) {
    gemm_bf16_kernel<float, FD_, HID_, 1><<<gProj, blk, 0, stream>>>(
        X, (long long)N_*FD_, 0LL,
        W, (long long)FD_*HID_,
        bias, outp, N_, FD_, HEADS_);
  };
  auto makeG = [&](const unsigned short* F2x, const float* ew) {
    gemm_bf16_kernel<unsigned short, HID_, 1, HID_><<<gProj, blk, 0, stream>>>(
        F2x, (long long)HEADS_*N_*HID_, (long long)N_*HID_,
        ew, 0LL,                         // W(l,k) = ew[k,l]  (ew^T access)
        nullptr, Gw, N_, HID_, HEADS_);
  };
  auto outp = [&](float* o, int acc) {
    outproj_kernel<<<gOut, blk, 0, stream>>>(Mw, Wout, bout, boutp, o, acc);
  };

  // per-head projections (shared by all 4 aggregates)
  proj(fa, W1, b1, F1a); proj(fa, W2, b2, F2a); proj(fa, W3, b3, F3a);
  proj(fb, W1, b1, F1b); proj(fb, W2, b2, F2b); proj(fb, W3, b3, F3b);

  float* out0 = (float*)d_out;
  float* out1 = out0 + (size_t)B_ * N_ * FD_;

  // ua: src=a,tgt=a, adj=A_intra_a, ew=W_intra_a
  makeG(F2a, Wia);
  attn_kernel<N_, 1><<<gAttn, blk, 0, stream>>>(F1a, Gw, F3a, Aia, Mw, HEADS_);
  outp(out0, 0);
  // ia: src=a,tgt=b, adj=A_inter, ew=W_inter
  makeG(F2b, Wit);
  attn_kernel<N_, 1><<<gAttn, blk, 0, stream>>>(F1a, Gw, F3a, Ait, Mw, HEADS_);
  outp(out0, 1);
  // ub: src=b,tgt=b, adj=A_intra_b, ew=W_intra_b
  makeG(F2b, Wib);
  attn_kernel<N_, 1><<<gAttn, blk, 0, stream>>>(F1b, Gw, F3b, Aib, Mw, HEADS_);
  outp(out1, 0);
  // ib: src=b,tgt=a, adj=A_inter^T, ew=W_inter
  makeG(F2a, Wit);
  attn_kernel<1, N_><<<gAttn, blk, 0, stream>>>(F1b, Gw, F3b, Ait, Mw, HEADS_);
  outp(out1, 1);
}
